// VQLayer_55551107006707
// MI455X (gfx1250) — matrix operand order repaired and checked by reading, not compiled, not run on hardware
//
#include <hip/hip_runtime.h>
#include <stdint.h>

typedef float    v4f  __attribute__((ext_vector_type(4)));
typedef float    v8f  __attribute__((ext_vector_type(8)));
typedef _Float16 v4h  __attribute__((ext_vector_type(4)));
typedef _Float16 v8h  __attribute__((ext_vector_type(8)));
typedef _Float16 v16h __attribute__((ext_vector_type(16)));
typedef unsigned int v4u __attribute__((ext_vector_type(4)));
typedef int      v4i  __attribute__((ext_vector_type(4)));
typedef int      v8i  __attribute__((ext_vector_type(8)));

#define D_DIM   256
#define K_CODES 8192
#define N_ROWS  16384
#define EH_STRIDE 264   // halves/row in LDS: 256 + 8 pad = 132 dwords -> conflict-free b128

// ---------------------------------------------------------------------------
// Kernel 1: squared L2 norms of the 8192 embedding rows (one wave per row).
// ---------------------------------------------------------------------------
__global__ __launch_bounds__(256) void vq_enorm_kernel(
    const float* __restrict__ emb, float* __restrict__ norms) {
  const int t    = threadIdx.x;
  const int row  = blockIdx.x * 8 + (t >> 5);
  const int lane = t & 31;
  const float* p = emb + (long long)row * D_DIM;
  v4f a = *(const v4f*)(p + lane * 4);
  v4f b = *(const v4f*)(p + 128 + lane * 4);
  float s = 0.f;
#pragma unroll
  for (int i = 0; i < 4; ++i) s += a[i] * a[i] + b[i] * b[i];
#pragma unroll
  for (int off = 16; off > 0; off >>= 1) s += __shfl_xor(s, off, 32);
  if (lane == 0) norms[row] = s;
}

// ---------------------------------------------------------------------------
// Kernel 1b: one-shot split of E into f16 hi/lo planes (row-major [8192][256]).
// ---------------------------------------------------------------------------
__global__ __launch_bounds__(256) void vq_esplit_kernel(
    const float* __restrict__ emb, _Float16* __restrict__ ehs,
    _Float16* __restrict__ els) {
  const long long i = ((long long)blockIdx.x * 256 + threadIdx.x) * 4;
  v4f v = *(const v4f*)(emb + i);
  v4h hh, ll;
#pragma unroll
  for (int j = 0; j < 4; ++j) {
    _Float16 h = (_Float16)v[j];
    hh[j] = h;
    ll[j] = (_Float16)(v[j] - (float)h);
  }
  *(v4h*)(ehs + i) = hh;
  *(v4h*)(els + i) = ll;
}

// ---------------------------------------------------------------------------
// Shared device helpers for the argmin kernels.
// ---------------------------------------------------------------------------
__device__ __forceinline__ void vq_load_a_split(
    const float* __restrict__ x, int rowBase, int col, int hi,
    v16h ah[8], v16h al[8]) {
  // 16-bit A layout (16x32 MxK): lane holds K = d0+16*hi+{0..15} (elems 0..15),
  // matching the contiguous B fill.
  const float* xrow = x + (long long)(rowBase + col) * D_DIM;
#pragma unroll
  for (int c = 0; c < 8; ++c) {
    const float* p = xrow + 32 * c + 16 * hi;
    v4f t0 = *(const v4f*)(p);
    v4f t1 = *(const v4f*)(p + 4);
    v4f t2 = *(const v4f*)(p + 8);
    v4f t3 = *(const v4f*)(p + 12);
#pragma unroll
    for (int j = 0; j < 4; ++j) {
      _Float16 h;
      h = (_Float16)t0[j]; ah[c][j]      = h; al[c][j]      = (_Float16)(t0[j] - (float)h);
      h = (_Float16)t1[j]; ah[c][4 + j]  = h; al[c][4 + j]  = (_Float16)(t1[j] - (float)h);
      h = (_Float16)t2[j]; ah[c][8 + j]  = h; al[c][8 + j]  = (_Float16)(t2[j] - (float)h);
      h = (_Float16)t3[j]; ah[c][12 + j] = h; al[c][12 + j] = (_Float16)(t3[j] - (float)h);
    }
  }
}

__device__ __forceinline__ void vq_subtile_wmma(
    const _Float16* __restrict__ bb, const _Float16* __restrict__ lb,
    const v16h ah[8], const v16h al[8], float en, int kc,
    float best[8], int bidx[8]) {
  v8f acc_hh = {0.f, 0.f, 0.f, 0.f, 0.f, 0.f, 0.f, 0.f};
  v8f acc_hl = acc_hh;
  v8f acc_lh = acc_hh;
#pragma unroll
  for (int c = 0; c < 8; ++c) {
    v8h h0 = *(const v8h*)(bb + 32 * c);
    v8h h1 = *(const v8h*)(bb + 32 * c + 8);
    v8h l0 = *(const v8h*)(lb + 32 * c);
    v8h l1 = *(const v8h*)(lb + 32 * c + 8);
    v16h bh = __builtin_shufflevector(h0, h1, 0, 1, 2, 3, 4, 5, 6, 7,
                                      8, 9, 10, 11, 12, 13, 14, 15);
    v16h bl = __builtin_shufflevector(l0, l1, 0, 1, 2, 3, 4, 5, 6, 7,
                                      8, 9, 10, 11, 12, 13, 14, 15);
    acc_hh = __builtin_amdgcn_wmma_f32_16x16x32_f16(
        false, ah[c], false, bh, (short)0, acc_hh, false, false);
    acc_hl = __builtin_amdgcn_wmma_f32_16x16x32_f16(
        false, ah[c], false, bl, (short)0, acc_hl, false, false);
    acc_lh = __builtin_amdgcn_wmma_f32_16x16x32_f16(
        false, al[c], false, bh, (short)0, acc_lh, false, false);
  }
#pragma unroll
  for (int r = 0; r < 8; ++r) {
    float dot  = acc_hh[r] + acc_hl[r] + acc_lh[r];
    float dist = fmaf(-2.f, dot, en);   // ||x||^2 irrelevant to argmin
    if (dist < best[r]) { best[r] = dist; bidx[r] = kc; }
  }
}

__device__ __forceinline__ void vq_reduce_store(
    float best[8], int bidx[8], int rowBase, int col, int hi,
    int* __restrict__ indices) {
#pragma unroll
  for (int r = 0; r < 8; ++r) {
    float d = best[r];
    int   i = bidx[r];
#pragma unroll
    for (int off = 1; off <= 8; off <<= 1) {
      float od = __shfl_xor(d, off, 32);
      int   oi = __shfl_xor(i, off, 32);
      if (od < d || (od == d && oi < i)) { d = od; i = oi; }
    }
    if (col == 0) indices[rowBase + 8 * hi + r] = i;   // C/D row = r + 8*hi
  }
}

// ---------------------------------------------------------------------------
// TDM descriptor: 2D tensor [8192 rows x 256 halves], tile [32 x 256], 2-byte
// elements, LDS padding 4 dwords per 128 dwords (-> 264-half LDS row stride).
// ISA 8.3/8.4 field packing; groups 2/3 zero (2-D tensor). 6-arg builtin form
// (clang-23 / therock): (v4u g0, v8i g1, v4i g2, v4i g3, v8i extra, i32 cpol).
// ---------------------------------------------------------------------------
__device__ __forceinline__ void vq_tdm_issue(
    const _Float16* gbase, int k0, unsigned lds_off) {
  unsigned long long ga =
      (unsigned long long)(uintptr_t)gbase + (unsigned long long)k0 * (D_DIM * 2);
  v4u g0;
  g0[0] = 1u;                                   // count=1, user descriptor
  g0[1] = lds_off;                              // lds_addr (bytes)
  g0[2] = (unsigned)(ga & 0xFFFFFFFFu);         // global_addr[31:0]
  g0[3] = (unsigned)((ga >> 32) & 0x01FFFFFFu)  // global_addr[56:32]
          | (2u << 30);                         // type=2 (image)
  v8i g1;
  g1[0] = (int)((1u << 16)      // data_size = 1 -> 2 bytes
                | (1u << 20)    // pad_enable
                | (6u << 22)    // pad_interval: 2^(6+1)=128 dwords
                | (3u << 25));  // pad_amount: 3+1 = 4 dwords
  g1[1] = (int)(256u << 16);                    // tensor_dim0 = 256 (lo16)
  g1[2] = (int)(8192u << 16);                   // dim0 hi=0 | tensor_dim1=8192
  g1[3] = (int)(256u << 16);                    // dim1 hi=0 | tile_dim0=256
  g1[4] = 32;                                   // tile_dim1=32 | tile_dim2=0
  g1[5] = 256;                                  // tensor_dim0_stride = 256
  g1[6] = 0;
  g1[7] = 0;
  v4i gz4 = {0, 0, 0, 0};
  v8i gz8 = {0, 0, 0, 0, 0, 0, 0, 0};
  __builtin_amdgcn_tensor_load_to_lds(g0, g1, gz4, gz4, gz8, 0);
}

// ---------------------------------------------------------------------------
// Kernel 2a (fast path): argmin with pre-split E planes staged by the Tensor
// Data Mover, double-buffered in LDS and overlapped with the WMMA stream.
// One barrier per 32-code tile; wave 0 drives the TDM + TENSORcnt gating.
// ---------------------------------------------------------------------------
__global__ __launch_bounds__(256) void vq_argmin_tdm_kernel(
    const float* __restrict__ x, const _Float16* __restrict__ ehs,
    const _Float16* __restrict__ els, const float* __restrict__ norms,
    int* __restrict__ indices) {
  __shared__ _Float16 EhBuf[2][32 * EH_STRIDE];
  __shared__ _Float16 ElBuf[2][32 * EH_STRIDE];

  const int tid  = threadIdx.x;
  const int wave = tid >> 5;
  const int lane = tid & 31;
  const int col  = lane & 15;
  const int hi   = lane >> 4;
  const int rowBase = blockIdx.x * 128 + wave * 16;
  const bool issuer = (wave == 0);

  v16h ah[8], al[8];
  vq_load_a_split(x, rowBase, col, hi, ah, al);

  float best[8];
  int   bidx[8];
#pragma unroll
  for (int r = 0; r < 8; ++r) { best[r] = 3.4e38f; bidx[r] = 0; }

  if (issuer) {
    vq_tdm_issue(ehs, 0, (unsigned)(uintptr_t)&EhBuf[0][0]);
    vq_tdm_issue(els, 0, (unsigned)(uintptr_t)&ElBuf[0][0]);
  }

  for (int k0 = 0; k0 < K_CODES; k0 += 32) {
    const int buf = (k0 >> 5) & 1;
    if (issuer) __builtin_amdgcn_s_wait_tensorcnt(0);  // tile k0 landed
    __syncthreads();  // tile visible to all; buf^1 free to overwrite
    if (issuer && (k0 + 32 < K_CODES)) {
      vq_tdm_issue(ehs, k0 + 32, (unsigned)(uintptr_t)&EhBuf[buf ^ 1][0]);
      vq_tdm_issue(els, k0 + 32, (unsigned)(uintptr_t)&ElBuf[buf ^ 1][0]);
    }
#pragma unroll
    for (int s = 0; s < 2; ++s) {
      const int erow = s * 16 + col;
      // B operand (32x16 KxN), contiguous k order as in A: lane holds 16 contiguous halves
      // e[col][d0 + 16*hi .. +15] -> two 16-byte LDS loads per operand.
      const _Float16* bb = &EhBuf[buf][erow * EH_STRIDE + 16 * hi];
      const _Float16* lb = &ElBuf[buf][erow * EH_STRIDE + 16 * hi];
      const int   kc = k0 + s * 16 + col;
      const float en = norms[kc];
      vq_subtile_wmma(bb, lb, ah, al, en, kc, best, bidx);
    }
  }

  vq_reduce_store(best, bidx, rowBase, col, hi, indices);
}

// ---------------------------------------------------------------------------
// Kernel 2b (fallback if ws too small for split-E planes): convert-in-kernel
// staging (round-2 version).
// ---------------------------------------------------------------------------
__global__ __launch_bounds__(256) void vq_argmin_conv_kernel(
    const float* __restrict__ x, const float* __restrict__ emb,
    const float* __restrict__ norms, int* __restrict__ indices) {
  __shared__ _Float16 Eh[32 * EH_STRIDE];
  __shared__ _Float16 El[32 * EH_STRIDE];
  __shared__ float    tileN[32];

  const int tid  = threadIdx.x;
  const int wave = tid >> 5;
  const int lane = tid & 31;
  const int col  = lane & 15;
  const int hi   = lane >> 4;
  const int rowBase = blockIdx.x * 128 + wave * 16;

  v16h ah[8], al[8];
  vq_load_a_split(x, rowBase, col, hi, ah, al);

  float best[8];
  int   bidx[8];
#pragma unroll
  for (int r = 0; r < 8; ++r) { best[r] = 3.4e38f; bidx[r] = 0; }

  for (int k0 = 0; k0 < K_CODES; k0 += 32) {
#pragma unroll
    for (int i = 0; i < 8; ++i) {
      int q    = tid + i * 256;
      int code = q >> 6;
      int d4   = (q & 63) * 4;
      v4f v = *(const v4f*)(emb + (long long)(k0 + code) * D_DIM + d4);
      v4h hh, ll;
#pragma unroll
      for (int j = 0; j < 4; ++j) {
        _Float16 h = (_Float16)v[j];
        hh[j] = h;
        ll[j] = (_Float16)(v[j] - (float)h);
      }
      *(v4h*)(&Eh[code * EH_STRIDE + d4]) = hh;
      *(v4h*)(&El[code * EH_STRIDE + d4]) = ll;
    }
    if (tid < 32) tileN[tid] = norms[k0 + tid];
    __syncthreads();

#pragma unroll
    for (int s = 0; s < 2; ++s) {
      const int erow = s * 16 + col;
      const _Float16* bb = &Eh[erow * EH_STRIDE + 16 * hi];
      const _Float16* lb = &El[erow * EH_STRIDE + 16 * hi];
      vq_subtile_wmma(bb, lb, ah, al, tileN[erow], k0 + s * 16 + col,
                      best, bidx);
    }
    __syncthreads();
  }

  vq_reduce_store(best, bidx, rowBase, col, hi, indices);
}

// ---------------------------------------------------------------------------
// Kernel 3: gather codes -> out (exact f32), per-block partials of (x-q)^2.
// ---------------------------------------------------------------------------
__global__ __launch_bounds__(256) void vq_gather_kernel(
    const float* __restrict__ x, const float* __restrict__ emb,
    const int* __restrict__ indices, float* __restrict__ out,
    float* __restrict__ partials) {
  __shared__ float red[256];
  const int t = threadIdx.x;
  const long long base = (long long)blockIdx.x * 1024 + t * 4;
  const int n = (int)(base >> 8);
  const int d = (int)(base & 255);
  const int idx = indices[n];
  v4f q   = *(const v4f*)(emb + (long long)idx * D_DIM + d);
  v4f xin = *(const v4f*)(x + base);
  *(v4f*)(out + base) = q;
  float s = 0.f;
#pragma unroll
  for (int i = 0; i < 4; ++i) { float df = xin[i] - q[i]; s += df * df; }
  red[t] = s;
  __syncthreads();
#pragma unroll
  for (int off = 128; off > 0; off >>= 1) {
    if (t < off) red[t] += red[t + off];
    __syncthreads();
  }
  if (t == 0) partials[blockIdx.x] = red[0];
}

// ---------------------------------------------------------------------------
// Kernel 4: final fixed-order reduction; loss = 1.25 * MSE.
// ---------------------------------------------------------------------------
__global__ __launch_bounds__(256) void vq_loss_kernel(
    const float* __restrict__ partials, float* __restrict__ loss_out) {
  __shared__ float red[256];
  const int t = threadIdx.x;
  float s = 0.f;
  for (int i = t; i < 4096; i += 256) s += partials[i];
  red[t] = s;
  __syncthreads();
#pragma unroll
  for (int off = 128; off > 0; off >>= 1) {
    if (t < off) red[t] += red[t + off];
    __syncthreads();
  }
  if (t == 0) loss_out[0] = red[0] * (1.25f / (float)(N_ROWS * D_DIM));
}

// ---------------------------------------------------------------------------
extern "C" void kernel_launch(void* const* d_in, const int* in_sizes, int n_in,
                              void* d_out, int out_size, void* d_ws, size_t ws_size,
                              hipStream_t stream) {
  const float* x   = (const float*)d_in[0];   // [16,32,32,256] f32
  const float* emb = (const float*)d_in[1];   // [8192,256] f32
  float* out = (float*)d_out;                  // 4194304 quantized + 1 loss

  char*  ws       = (char*)d_ws;
  float* norms    = (float*)ws;                                   // 32 KB
  int*   indices  = (int*)(ws + 32768);                           // 64 KB
  float* partials = (float*)(ws + 98304);                         // 16 KB
  _Float16* ehs   = (_Float16*)(ws + 131072);                     // 4 MB
  _Float16* els   = (_Float16*)(ws + 131072 + (size_t)K_CODES * D_DIM * 2); // 4 MB
  const size_t need_fast = 131072 + 2 * (size_t)K_CODES * D_DIM * 2;

  vq_enorm_kernel<<<K_CODES / 8, 256, 0, stream>>>(emb, norms);

  if (ws_size >= need_fast) {
    vq_esplit_kernel<<<(K_CODES * D_DIM) / 1024, 256, 0, stream>>>(emb, ehs, els);
    vq_argmin_tdm_kernel<<<N_ROWS / 128, 256, 0, stream>>>(x, ehs, els, norms,
                                                           indices);
  } else {
    vq_argmin_conv_kernel<<<N_ROWS / 128, 256, 0, stream>>>(x, emb, norms,
                                                            indices);
  }

  vq_gather_kernel<<<(N_ROWS * D_DIM) / 1024, 256, 0, stream>>>(x, emb, indices,
                                                                out, partials);
  vq_loss_kernel<<<1, 256, 0, stream>>>(partials, out + (long long)N_ROWS * D_DIM);
}
